// ResidualQuantizer_79259326480567
// MI455X (gfx1250) — matrix operand order repaired and checked by reading, not compiled, not run on hardware
//
#include <hip/hip_runtime.h>
#include <hip/hip_bf16.h>

typedef __attribute__((ext_vector_type(16))) __bf16 v16bf;
typedef __attribute__((ext_vector_type(8)))  float  v8f;

#define NTOK 16384       // B*T
#define DIM  256
#define KCB  4096
#define QST  4

// workspace layout (bytes)
#define WS_ZN   0u                       // 16384*256*2   = 8388608
#define WS_IDX  (8388608u)               // 16384*4*4     = 262144
#define WS_PART (WS_IDX + 262144u)       // 2048*4        = 8192
#define WS_CB   (WS_PART + 8192u)        // 4*4096*256*2  = 8388608

struct __attribute__((aligned(16))) U128 { unsigned int a, b, c, d; };

// ----------------------------------------------------------- async helpers
__device__ __forceinline__ unsigned lds_byte_offset(const void* p) {
    // generic -> LDS address space; ptrtoint of as(3) pointer = LDS byte offset
    return (unsigned)(unsigned long long)
        (__attribute__((address_space(3))) const void*)p;
}

__device__ __forceinline__ void async_cp16(unsigned lds_off, const char* g) {
    asm volatile("global_load_async_to_lds_b128 %0, %1, off"
                 :: "v"(lds_off), "v"(g) : "memory");
}

__device__ __forceinline__ void wait_async_le8() {
    asm volatile("s_wait_asynccnt 0x8" ::: "memory");
}
__device__ __forceinline__ void wait_async_0() {
    asm volatile("s_wait_asynccnt 0x0" ::: "memory");
}

// ---------------------------------------------------------------- normalize
__global__ __launch_bounds__(256) void k_normalize(const float* __restrict__ x,
                                                   unsigned short* __restrict__ zn) {
    const int tok  = (blockIdx.x * 256 + threadIdx.x) >> 5;   // one wave32 / token
    const int lane = threadIdx.x & 31;
    const float* row = x + (size_t)tok * DIM + lane * 8;
    float v[8];
    #pragma unroll
    for (int j = 0; j < 8; ++j) v[j] = row[j];
    float s = 0.f;
    #pragma unroll
    for (int j = 0; j < 8; ++j) s += v[j] * v[j];
    #pragma unroll
    for (int off = 16; off >= 1; off >>= 1) s += __shfl_xor(s, off, 32);
    const float scale = 1.0f / fmaxf(sqrtf(s), 1e-12f);
    __bf16 h[8];
    #pragma unroll
    for (int j = 0; j < 8; ++j) h[j] = (__bf16)(v[j] * scale);
    __builtin_memcpy(zn + (size_t)tok * DIM + lane * 8, h, 16);
}

// ---------------------------------------------------- codebooks f32 -> bf16
__global__ __launch_bounds__(256) void k_cvt_cb(const float* __restrict__ cb,
                                                unsigned short* __restrict__ cbh) {
    const size_t i = ((size_t)blockIdx.x * 256 + threadIdx.x) * 8;
    if (i >= (size_t)QST * KCB * DIM) return;
    float v[8];
    #pragma unroll
    for (int j = 0; j < 8; ++j) v[j] = cb[i + j];
    __bf16 h[8];
    #pragma unroll
    for (int j = 0; j < 8; ++j) h[j] = (__bf16)v[j];
    __builtin_memcpy(cbh + i, h, 16);
}

// ------------------------------------------- WMMA GEMM + streaming argmax
// grid: (64 token tiles, 4 stages), block 256 (8 waves); wave owns 32 tokens.
// Codebook chunks (64 codes) double-buffered in LDS via async-to-LDS DMA.
#define LDS_PITCH 264                     // 528B rows: conflict-free b128 reads
#define LDS_BUF_SH (64 * LDS_PITCH)       // ushorts per buffer
#define LDS_BUF_BYTES (LDS_BUF_SH * 2)    // 33792 B
__global__ __launch_bounds__(256) void k_argmax(const unsigned short* __restrict__ znh,
                                                const unsigned short* __restrict__ cbh,
                                                int* __restrict__ idx_out) {
    __shared__ unsigned short lds[2 * LDS_BUF_SH];
    const int q       = blockIdx.y;
    const int lane    = threadIdx.x & 31;
    const int wv      = threadIdx.x >> 5;
    const int tokBase = blockIdx.x * 256 + wv * 32;
    const int hi      = (lane >> 4) & 1;
    const unsigned short* cb = cbh + (size_t)q * KCB * DIM;

    // A fragments: two 16-token sets (K=256 -> 8 frags each) resident in VGPRs.
    // ISA 16-bit A layout: lanes 0-15: K={k0..+7, k0+16..+23}; lanes 16-31: {+8..+15,+24..+31}
    v16bf afrag[16];
    #pragma unroll
    for (int s = 0; s < 2; ++s) {
        const char* rowp =
            (const char*)(znh + (size_t)(tokBase + s * 16 + (lane & 15)) * DIM);
        #pragma unroll
        for (int kf = 0; kf < 8; ++kf) {
            const char* p = rowp + kf * 64 + hi * 16;
            __builtin_memcpy(&afrag[s * 8 + kf], p, 16);
            __builtin_memcpy(((char*)&afrag[s * 8 + kf]) + 16, p + 32, 16);
        }
    }

    float best[16]; int bidx[16];
    #pragma unroll
    for (int r = 0; r < 16; ++r) { best[r] = -3.0e38f; bidx[r] = 0; }

    const int crow  = threadIdx.x >> 2;    // 0..63: codebook row within chunk
    const int cquad = threadIdx.x & 3;     // 128B slice of the 512B bf16 row
    const unsigned ldsBase = lds_byte_offset(lds);

    // ---- async double-buffered pipeline over 64 chunks of 64 codes ----
    {
        const char* g0 = (const char*)(cb + (size_t)crow * DIM) + cquad * 128;
        const unsigned l0 = ldsBase + crow * (LDS_PITCH * 2) + cquad * 128;
        #pragma unroll
        for (int j = 0; j < 8; ++j) async_cp16(l0 + j * 16, g0 + j * 16);
    }

    int buf = 0;
    for (int chunk = 0; chunk < KCB / 64; ++chunk) {
        const bool haveNext = (chunk + 1) < (KCB / 64);
        if (haveNext) {   // prefetch next chunk into other buffer (overlaps compute wait)
            const char* g = (const char*)(cb + (size_t)((chunk + 1) * 64 + crow) * DIM)
                          + cquad * 128;
            const unsigned l = ldsBase + (buf ^ 1) * LDS_BUF_BYTES
                             + crow * (LDS_PITCH * 2) + cquad * 128;
            #pragma unroll
            for (int j = 0; j < 8; ++j) async_cp16(l + j * 16, g + j * 16);
        }
        // async loads retire in order: <=8 outstanding => current chunk landed
        if (haveNext) wait_async_le8(); else wait_async_0();
        __syncthreads();

        const char* cbuf = (const char*)lds + buf * LDS_BUF_BYTES;
        #pragma unroll
        for (int nt = 0; nt < 4; ++nt) {
            v8f acc0 = {};
            v8f acc1 = {};
            // B layout 32x16 bf16: lane holds col = lane%16, K={k0+8h..+7, k0+16+8h..+7}
            const char* bbase = cbuf + (nt * 16 + (lane & 15)) * (LDS_PITCH * 2)
                              + hi * 16;
            #pragma unroll
            for (int kf = 0; kf < 8; ++kf) {
                v16bf bf;
                __builtin_memcpy(&bf, bbase + kf * 64, 16);
                __builtin_memcpy(((char*)&bf) + 16, bbase + kf * 64 + 32, 16);
                acc0 = __builtin_amdgcn_wmma_f32_16x16x32_bf16(
                        false, afrag[kf],     false, bf, (short)0, acc0, false, false);
                acc1 = __builtin_amdgcn_wmma_f32_16x16x32_bf16(
                        false, afrag[8 + kf], false, bf, (short)0, acc1, false, false);
            }
            const int code = chunk * 64 + nt * 16 + (lane & 15);
            #pragma unroll
            for (int r = 0; r < 8; ++r) {
                const float v0 = acc0[r];
                if (v0 > best[r])     { best[r] = v0;     bidx[r] = code; }
                const float v1 = acc1[r];
                if (v1 > best[8 + r]) { best[8 + r] = v1; bidx[8 + r] = code; }
            }
        }
        __syncthreads();   // compute done before this buffer gets overwritten
        buf ^= 1;
    }

    // cross-lane argmax within 16-lane groups (C layout: vgpr r -> row r / r+8)
    #pragma unroll
    for (int r = 0; r < 16; ++r) {
        #pragma unroll
        for (int off = 8; off >= 1; off >>= 1) {
            const float ov = __shfl_xor(best[r], off, 32);
            const int   oi = __shfl_xor(bidx[r], off, 32);
            if (ov > best[r] || (ov == best[r] && oi < bidx[r])) { best[r] = ov; bidx[r] = oi; }
        }
    }
    if (lane == 0) {
        #pragma unroll
        for (int r = 0; r < 8; ++r) {
            idx_out[(size_t)(tokBase + r) * QST + q]      = bidx[r];
            idx_out[(size_t)(tokBase + 16 + r) * QST + q] = bidx[8 + r];
        }
    } else if (lane == 16) {
        #pragma unroll
        for (int r = 0; r < 8; ++r) {
            idx_out[(size_t)(tokBase + 8 + r) * QST + q]  = bidx[r];
            idx_out[(size_t)(tokBase + 24 + r) * QST + q] = bidx[8 + r];
        }
    }
}

// -------------------------------------------- gather + sum + loss partials
__global__ __launch_bounds__(256) void k_finalize(const float* __restrict__ x,
                                                  const float* __restrict__ cbf,
                                                  const int* __restrict__ idx,
                                                  float* __restrict__ out,
                                                  float* __restrict__ partial) {
    __shared__ float wsum[8];
    const int wv   = threadIdx.x >> 5;
    const int lane = threadIdx.x & 31;
    const int tok  = blockIdx.x * 8 + wv;

    const float* row = x + (size_t)tok * DIM + lane * 8;
    float v[8];
    #pragma unroll
    for (int j = 0; j < 8; ++j) v[j] = row[j];
    float s = 0.f;
    #pragma unroll
    for (int j = 0; j < 8; ++j) s += v[j] * v[j];
    #pragma unroll
    for (int off = 16; off >= 1; off >>= 1) s += __shfl_xor(s, off, 32);
    const float scale = 1.0f / fmaxf(sqrtf(s), 1e-12f);

    float zn[8], acc[8];
    #pragma unroll
    for (int j = 0; j < 8; ++j) { zn[j] = v[j] * scale; acc[j] = 0.f; }

    float lloss = 0.f;
    #pragma unroll
    for (int q = 0; q < QST; ++q) {
        const int id = idx[(size_t)tok * QST + q];
        const float* e = cbf + ((size_t)q * KCB + id) * DIM + lane * 8;
        #pragma unroll
        for (int j = 0; j < 8; ++j) {
            const float ev = e[j];
            acc[j] += ev;
            const float d = ev - zn[j];
            lloss += d * d;
        }
    }
    float* orow = out + (size_t)tok * DIM + lane * 8;
    #pragma unroll
    for (int j = 0; j < 8; ++j) orow[j] = acc[j];

    if (lane < QST)   // indices as float, token-major (N, Q)
        out[(size_t)NTOK * DIM + 1 + (size_t)tok * QST + lane] =
            (float)idx[(size_t)tok * QST + lane];

    #pragma unroll
    for (int off = 16; off >= 1; off >>= 1) lloss += __shfl_xor(lloss, off, 32);
    if (lane == 0) wsum[wv] = lloss;
    __syncthreads();
    if (threadIdx.x == 0) {
        float t = 0.f;
        #pragma unroll
        for (int w = 0; w < 8; ++w) t += wsum[w];
        partial[blockIdx.x] = t;
    }
}

__global__ __launch_bounds__(256) void k_loss_reduce(const float* __restrict__ partial,
                                                     float* __restrict__ out) {
    __shared__ float sm[256];
    float s = 0.f;
    for (int i = threadIdx.x; i < 2048; i += 256) s += partial[i];
    sm[threadIdx.x] = s;
    __syncthreads();
    for (int st = 128; st >= 1; st >>= 1) {
        if (threadIdx.x < st) sm[threadIdx.x] += sm[threadIdx.x + st];
        __syncthreads();
    }
    if (threadIdx.x == 0)
        out[(size_t)NTOK * DIM] = sm[0] / (float)((size_t)NTOK * DIM);
}

// ------------------------------------------------------------------ launch
extern "C" void kernel_launch(void* const* d_in, const int* in_sizes, int n_in,
                              void* d_out, int out_size, void* d_ws, size_t ws_size,
                              hipStream_t stream) {
    const float* x  = (const float*)d_in[0];      // (16,1024,256) f32
    const float* cb = (const float*)d_in[1];      // (4,4096,256) f32, rows l2-normed
    float* out = (float*)d_out;
    char*  ws  = (char*)d_ws;

    unsigned short* zn   = (unsigned short*)(ws + WS_ZN);
    int*            idx  = (int*)(ws + WS_IDX);
    float*          part = (float*)(ws + WS_PART);
    unsigned short* cbh  = (unsigned short*)(ws + WS_CB);

    k_normalize<<<NTOK / 8, 256, 0, stream>>>(x, zn);
    k_cvt_cb<<<(QST * KCB * DIM) / (256 * 8), 256, 0, stream>>>(cb, cbh);

    dim3 g(NTOK / 256, QST);
    k_argmax<<<g, 256, 0, stream>>>(zn, cbh, idx);

    k_finalize<<<NTOK / 8, 256, 0, stream>>>(x, cb, idx, out, part);
    k_loss_reduce<<<1, 256, 0, stream>>>(part, out);
}
